// MultiHeadAtt_5317169513201
// MI455X (gfx1250) — matrix operand order repaired and checked by reading, not compiled, not run on hardware
//
#include <hip/hip_runtime.h>
#include <hip/hip_bf16.h>

typedef __attribute__((ext_vector_type(16))) _Float16 v16h;
typedef __attribute__((ext_vector_type(8)))  _Float16 v8h;
typedef __attribute__((ext_vector_type(4)))  _Float16 v4h;
typedef __attribute__((ext_vector_type(8)))  float    v8f;
typedef int b128_t __attribute__((vector_size(16)));   // matches builtin pointee

#define H_   16
#define D_   1024
#define DQ_  64
#define B_   2
#define S_   2048
#define BS_  4096

#if defined(__AMDGCN__) && __has_builtin(__builtin_amdgcn_global_load_async_to_lds_b128)
#define HAVE_ASYNC_LDS 1
#endif

// Async global->LDS 16B copy (ASYNCcnt-tracked, no VGPR round-trip) with a
// sync fallback if the builtin is absent on this toolchain.
__device__ __forceinline__ void async_ld_b128(const _Float16* g, _Float16* l) {
#ifdef HAVE_ASYNC_LDS
  __builtin_amdgcn_global_load_async_to_lds_b128(
      (__attribute__((address_space(1))) b128_t*)g,
      (__attribute__((address_space(3))) b128_t*)l, 0, 0);
#else
  *(v8h*)l = *(const v8h*)g;
#endif
}
// Wait until <= n async ops outstanding for this wave.
__device__ __forceinline__ void wait_async_le8() {
#ifdef HAVE_ASYNC_LDS
  asm volatile("s_wait_asynccnt 0x8" ::: "memory");
#endif
}
__device__ __forceinline__ void wait_async_0() {
#ifdef HAVE_ASYNC_LDS
  asm volatile("s_wait_asynccnt 0x0" ::: "memory");
#endif
}

__device__ __forceinline__ v8f wmma_f32_f16(v16h a, v16h b, v8f c) {
  // D = A(16x32 f16) * B(32x16 f16) + C(16x16 f32)
  return __builtin_amdgcn_wmma_f32_16x16x32_f16(false, a, false, b, (short)0, c,
                                                false, false);
}

// A-operand fragment (16x32, M x K). Caller passes p = rowBase + akb where
// akb = (lane<16) ? 0 : 8.  v16h elems 0..7 = K kb..kb+7, elems 8..15 = K kb+16..kb+23
// per ISA 7.12.2 (16-bit A-matrix layout).
__device__ __forceinline__ v16h frag_a(const _Float16* p) {
  v8h lo = *(const v8h*)p;
  v8h hi = *(const v8h*)(p + 16);
  v16h r;
#pragma unroll
  for (int i = 0; i < 8; ++i) { r[i] = lo[i]; r[i + 8] = hi[i]; }
  return r;
}

// B-operand fragment (32x16, K x N) from a [N,K] row-major tile. Caller passes
// p = colRowBase + bkb where bkb = (lane<16) ? 0 : 8. elems 0..7 = K kb..kb+7, 8..15 = K kb+16..kb+23.
__device__ __forceinline__ v16h frag_b(const _Float16* p) {
  v8h lo = *(const v8h*)p;
  v8h hi = *(const v8h*)(p + 16);
  v16h r;
#pragma unroll
  for (int i = 0; i < 8; ++i) { r[i] = lo[i]; r[i + 8] = hi[i]; }
  return r;
}

// ---------------------------------------------------------------- f32 -> f16
__global__ void cvt_f16(const float* __restrict__ src, _Float16* __restrict__ dst,
                        int n4) {
  int i = blockIdx.x * blockDim.x + threadIdx.x;
  if (i >= n4) return;
  float4 v = reinterpret_cast<const float4*>(src)[i];
  v4h o;
  o[0] = (_Float16)v.x; o[1] = (_Float16)v.y;
  o[2] = (_Float16)v.z; o[3] = (_Float16)v.w;
  reinterpret_cast<v4h*>(dst)[i] = o;
}

// ---------------------------------------- f32 transpose + cvt (dst[j,i]=src[i,j])
__global__ void transpose_cvt(const float* __restrict__ src,
                              _Float16* __restrict__ dst, int dim) {
  __shared__ float tile[32][33];
  int bx = blockIdx.x * 32, by = blockIdx.y * 32;
  int tx = threadIdx.x, ty = threadIdx.y;  // 32 x 8
#pragma unroll
  for (int r = 0; r < 32; r += 8)
    tile[ty + r][tx] = src[(size_t)(by + ty + r) * dim + bx + tx];
  __syncthreads();
#pragma unroll
  for (int r = 0; r < 32; r += 8)
    dst[(size_t)(bx + ty + r) * dim + by + tx] = (_Float16)tile[tx][ty + r];
}

// ------------------------------------------------------------------ WMMA GEMM
// C[M,N] = A[M,K] * Bt[N,K]^T   (row-major, K innermost on both operands)
// 128x128 block, K-slab 64, double-buffered LDS filled by async global->LDS.
// MODE 0: f16 row-major out.  MODE 1: f16 scatter into [B,H,S,dq].  MODE 2: f32 out.
template <int MODE>
__global__ void __launch_bounds__(256)
gemm_nt(const _Float16* __restrict__ A, const _Float16* __restrict__ Bt,
        void* __restrict__ Cout, int M, int N, int K) {
  __shared__ __align__(16) _Float16 As[2][128 * 64];
  __shared__ __align__(16) _Float16 Bs[2][128 * 64];
  const int tid  = threadIdx.x;
  const int lane = tid & 31;
  const int wid  = tid >> 5;   // 8 waves
  const int wm   = wid & 3;    // 4 M-groups of 32 rows
  const int wn   = wid >> 2;   // 2 N-groups of 64 cols
  const int bm   = blockIdx.y * 128;
  const int bn   = blockIdx.x * 128;
  const int lrow = lane & 15;
  const int akb  = (lane < 16) ? 0 : 8;
  const int bkb  = (lane < 16) ? 0 : 8;

  v8f acc[2][4];
#pragma unroll
  for (int mi = 0; mi < 2; ++mi)
#pragma unroll
    for (int ni = 0; ni < 4; ++ni)
#pragma unroll
      for (int i = 0; i < 8; ++i) acc[mi][ni][i] = 0.f;

  // per-thread load map: 1024 16B chunks per (A,B) slab pair, 256 threads x 4
  auto issue = [&](int buf, int k0) {
#pragma unroll
    for (int j = 0; j < 4; ++j) {
      int c   = tid + 256 * j;
      int r   = c >> 3;          // 0..127
      int col = (c & 7) * 8;     // 0..56
      async_ld_b128(A  + (size_t)(bm + r) * K + k0 + col, &As[buf][r * 64 + col]);
      async_ld_b128(Bt + (size_t)(bn + r) * K + k0 + col, &Bs[buf][r * 64 + col]);
    }
  };

  const int nk = K >> 6;
  issue(0, 0);
  for (int it = 0; it < nk; ++it) {
    const int cur = it & 1;
    if (it + 1 < nk) {
      issue(cur ^ 1, (it + 1) << 6);  // prefetch next slab under this one
      wait_async_le8();               // slab `it` landed; 8 new still in flight
    } else {
      wait_async_0();
    }
    __syncthreads();
#pragma unroll
    for (int kc = 0; kc < 2; ++kc) {
      v16h af[2], bf[4];
#pragma unroll
      for (int mi = 0; mi < 2; ++mi)
        af[mi] = frag_a(&As[cur][(wm * 32 + mi * 16 + lrow) * 64 + kc * 32 + akb]);
#pragma unroll
      for (int ni = 0; ni < 4; ++ni)
        bf[ni] = frag_b(&Bs[cur][(wn * 64 + ni * 16 + lrow) * 64 + kc * 32 + bkb]);
#pragma unroll
      for (int mi = 0; mi < 2; ++mi)
#pragma unroll
        for (int ni = 0; ni < 4; ++ni)
          acc[mi][ni] = wmma_f32_f16(af[mi], bf[ni], acc[mi][ni]);
    }
    __syncthreads();  // all waves done reading buf `cur` before it is refilled
  }

  const int hi8 = (lane < 16) ? 0 : 8;
#pragma unroll
  for (int mi = 0; mi < 2; ++mi)
#pragma unroll
    for (int ni = 0; ni < 4; ++ni)
#pragma unroll
      for (int i = 0; i < 8; ++i) {  // C layout: VGPR i -> row i (+8 hi half)
        int m = bm + wm * 32 + mi * 16 + i + hi8;
        int n = bn + wn * 64 + ni * 16 + lrow;
        float v = acc[mi][ni][i];
        if (MODE == 0) {
          ((_Float16*)Cout)[(size_t)m * N + n] = (_Float16)v;
        } else if (MODE == 1) {
          int b = m >> 11, s = m & (S_ - 1), h = n >> 6, e = n & (DQ_ - 1);
          ((_Float16*)Cout)[(((size_t)(b * H_ + h)) * S_ + s) * DQ_ + e] = (_Float16)v;
        } else {
          ((float*)Cout)[(size_t)m * N + n] = v;
        }
      }
}

// ----------------------------------------------------- causal flash attention
// grid.x = (b*H + h)*32 + rowTile ; 128 threads = 4 waves, 16 rows per wave.
__global__ void __launch_bounds__(128)
attn_causal(const _Float16* __restrict__ Q, const _Float16* __restrict__ Kmat,
            const _Float16* __restrict__ Vmat, _Float16* __restrict__ O) {
  __shared__ __align__(16) _Float16 Ks[64 * 64];       // [key][e]
  __shared__ __align__(16) _Float16 Vt[64 * 64];       // [e][key] (transposed)
  __shared__ __align__(16) _Float16 Ps[4 * 16 * 64];   // per-wave P tile
  const int bh   = blockIdx.x >> 5;   // b*16 + h
  const int rt   = blockIdx.x & 31;
  const int b    = bh >> 4, h = bh & 15;
  const int tid  = threadIdx.x;
  const int lane = tid & 31;
  const int w    = tid >> 5;
  const _Float16* qb = Q    + (size_t)bh * S_ * DQ_;
  const _Float16* kp = Kmat + (size_t)bh * S_ * DQ_;
  const _Float16* vp = Vmat + (size_t)bh * S_ * DQ_;
  const int lrow = lane & 15;
  const int akb  = (lane < 16) ? 0 : 8;
  const int bkb  = (lane < 16) ? 0 : 8;
  const int hi8  = (lane < 16) ? 0 : 8;
  const int r0   = rt * 64 + w * 16;

  v16h qf[2];  // resident Q fragments (K = d_qkv = 64 -> two K=32 chunks)
#pragma unroll
  for (int c = 0; c < 2; ++c)
    qf[c] = frag_a(qb + (size_t)(r0 + lrow) * DQ_ + c * 32 + akb);

  v8f oacc[4];
  float rmax[8], rsum[8];
#pragma unroll
  for (int t = 0; t < 4; ++t)
#pragma unroll
    for (int i = 0; i < 8; ++i) oacc[t][i] = 0.f;
#pragma unroll
  for (int i = 0; i < 8; ++i) { rmax[i] = -1e30f; rsum[i] = 0.f; }

  _Float16* Pw = Ps + w * 16 * 64;

  for (int kt = 0; kt <= rt; ++kt) {
    __syncthreads();
#pragma unroll
    for (int j = 0; j < 4; ++j) {  // 512 x 16B chunks, 128 threads
      int c   = tid + 128 * j;
      int row = c >> 3;
      int col = (c & 7) * 8;
      // K tile: async straight into LDS
      async_ld_b128(kp + (size_t)(kt * 64 + row) * DQ_ + col, Ks + row * 64 + col);
      // V tile: through VGPRs so it can be transposed on the way in
      v8h vv = *(const v8h*)(vp + (size_t)(kt * 64 + row) * DQ_ + col);
#pragma unroll
      for (int i = 0; i < 8; ++i) Vt[(col + i) * 64 + row] = vv[i];
    }
    wait_async_0();
    __syncthreads();

    // scores: S = Q (16x64) * K^T (64x64) -> 4 N-tiles, 2 WMMAs each
    v8f sc[4];
#pragma unroll
    for (int ni = 0; ni < 4; ++ni) {
      v8f s;
#pragma unroll
      for (int i = 0; i < 8; ++i) s[i] = 0.f;
      const _Float16* kr = Ks + (ni * 16 + lrow) * 64;
      s = wmma_f32_f16(qf[0], frag_b(kr + 0 + bkb), s);
      s = wmma_f32_f16(qf[1], frag_b(kr + 32 + bkb), s);
      sc[ni] = s;
    }

    // scale + causal mask on the diagonal block
    const bool diag = (kt == rt);
#pragma unroll
    for (int ni = 0; ni < 4; ++ni) {
      int col = kt * 64 + ni * 16 + lrow;
#pragma unroll
      for (int i = 0; i < 8; ++i) {
        float sv  = sc[ni][i] * 0.125f;  // 1/sqrt(64)
        int   row = r0 + i + hi8;
        sc[ni][i] = (diag && col > row) ? -1e30f : sv;
      }
    }

    // online softmax: each row lives in one VGPR across a 16-lane half
#pragma unroll
    for (int i = 0; i < 8; ++i) {
      float m = fmaxf(fmaxf(sc[0][i], sc[1][i]), fmaxf(sc[2][i], sc[3][i]));
#pragma unroll
      for (int d = 1; d < 16; d <<= 1) m = fmaxf(m, __shfl_xor(m, d, 32));
      float nm    = fmaxf(rmax[i], m);
      float alpha = __expf(rmax[i] - nm);
      rmax[i]     = nm;
      float ls = 0.f;
#pragma unroll
      for (int t = 0; t < 4; ++t) {
        float pij = __expf(sc[t][i] - nm);
        sc[t][i]  = pij;
        ls += pij;
      }
#pragma unroll
      for (int d = 1; d < 16; d <<= 1) ls += __shfl_xor(ls, d, 32);
      rsum[i] = rsum[i] * alpha + ls;
#pragma unroll
      for (int t = 0; t < 4; ++t) oacc[t][i] *= alpha;
    }

    // re-layout P (C-layout f32) -> A-layout f16 via per-wave LDS tile
#pragma unroll
    for (int ni = 0; ni < 4; ++ni)
#pragma unroll
      for (int i = 0; i < 8; ++i)
        Pw[(i + hi8) * 64 + ni * 16 + lrow] = (_Float16)sc[ni][i];

    v16h pf[2];
#pragma unroll
    for (int c = 0; c < 2; ++c)
      pf[c] = frag_a(Pw + lrow * 64 + c * 32 + akb);

    // O += P (16x64) * V (64x64); Vt is [e][key] so B-fragments are contiguous
#pragma unroll
    for (int c = 0; c < 2; ++c)
#pragma unroll
      for (int ei = 0; ei < 4; ++ei)
        oacc[ei] = wmma_f32_f16(
            pf[c], frag_b(Vt + (ei * 16 + lrow) * 64 + c * 32 + bkb), oacc[ei]);
  }

#pragma unroll
  for (int i = 0; i < 8; ++i) rsum[i] = 1.f / rsum[i];
#pragma unroll
  for (int ei = 0; ei < 4; ++ei)
#pragma unroll
    for (int i = 0; i < 8; ++i) {
      int srow = r0 + i + hi8;
      O[((size_t)(b * S_ + srow)) * D_ + h * DQ_ + ei * 16 + lrow] =
          (_Float16)(oacc[ei][i] * rsum[i]);
    }
}

// ---------------------------------------------------------------------- launch
extern "C" void kernel_launch(void* const* d_in, const int* in_sizes, int n_in,
                              void* d_out, int out_size, void* d_ws, size_t ws_size,
                              hipStream_t stream) {
  (void)in_sizes; (void)n_in; (void)out_size; (void)ws_size;
  const float* E  = (const float*)d_in[0];
  // d_in[1] = causal mask (tril) -- applied analytically in attn_causal
  const float* Wq = (const float*)d_in[2];
  const float* Wk = (const float*)d_in[3];
  const float* Wv = (const float*)d_in[4];
  const float* Hq = (const float*)d_in[5];
  const float* Hk = (const float*)d_in[6];
  const float* Hv = (const float*)d_in[7];
  const float* Wo = (const float*)d_in[8];

  const size_t MB = 1ull << 20;
  char* p = (char*)d_ws;
  _Float16* Eh  = (_Float16*)(p + 0);        // 8 MB, reused as attention output
  _Float16* HqH = (_Float16*)(p + 8 * MB);
  _Float16* HkH = (_Float16*)(p + 10 * MB);
  _Float16* HvH = (_Float16*)(p + 12 * MB);
  _Float16* WqT = (_Float16*)(p + 14 * MB);
  _Float16* WkT = (_Float16*)(p + 16 * MB);
  _Float16* WvT = (_Float16*)(p + 18 * MB);
  _Float16* Cq  = (_Float16*)(p + 20 * MB);
  _Float16* Ck  = (_Float16*)(p + 22 * MB);
  _Float16* Cv  = (_Float16*)(p + 24 * MB);
  _Float16* Qh  = (_Float16*)(p + 26 * MB);  // 8 MB [B,H,S,dq]
  _Float16* Kh  = (_Float16*)(p + 34 * MB);
  _Float16* Vh  = (_Float16*)(p + 42 * MB);
  _Float16* WoH = (_Float16*)(p + 50 * MB);
  _Float16* Oh  = Eh;  // E is dead once Q/K/V are projected

  int n4 = BS_ * D_ / 4;
  cvt_f16<<<(n4 + 255) / 256, 256, 0, stream>>>(E, Eh, n4);
  n4 = D_ * D_ / 4;
  cvt_f16<<<(n4 + 255) / 256, 256, 0, stream>>>(Hq, HqH, n4);
  cvt_f16<<<(n4 + 255) / 256, 256, 0, stream>>>(Hk, HkH, n4);
  cvt_f16<<<(n4 + 255) / 256, 256, 0, stream>>>(Hv, HvH, n4);
  cvt_f16<<<(n4 + 255) / 256, 256, 0, stream>>>(Wo, WoH, n4);

  dim3 tb(32, 8), tg(32, 32);
  transpose_cvt<<<tg, tb, 0, stream>>>(Wq, WqT, D_);
  transpose_cvt<<<tg, tb, 0, stream>>>(Wk, WkT, D_);
  transpose_cvt<<<tg, tb, 0, stream>>>(Wv, WvT, D_);

  // Combined weights: Cx[n,d] = sum_m Hx[n,m] * Wx[m,d]  (q = E * Cx^T)
  gemm_nt<0><<<dim3(8, 8), 256, 0, stream>>>(HqH, WqT, Cq, 1024, 1024, 1024);
  gemm_nt<0><<<dim3(8, 8), 256, 0, stream>>>(HkH, WkT, Ck, 1024, 1024, 1024);
  gemm_nt<0><<<dim3(8, 8), 256, 0, stream>>>(HvH, WvT, Cv, 1024, 1024, 1024);

  // Fused QKV head projections straight from E, scattered to [B,H,S,dq]
  gemm_nt<1><<<dim3(8, 32), 256, 0, stream>>>(Eh, Cq, Qh, BS_, 1024, 1024);
  gemm_nt<1><<<dim3(8, 32), 256, 0, stream>>>(Eh, Ck, Kh, BS_, 1024, 1024);
  gemm_nt<1><<<dim3(8, 32), 256, 0, stream>>>(Eh, Cv, Vh, BS_, 1024, 1024);

  attn_causal<<<B_ * H_ * (S_ / 64), 128, 0, stream>>>(Qh, Kh, Vh, Oh);

  // Final projection, f32 output
  gemm_nt<2><<<dim3(8, 32), 256, 0, stream>>>(Oh, WoH, d_out, BS_, 1024, 1024);
}
